// PaiNN_39307540693932
// MI455X (gfx1250) — matrix operand order repaired and checked by reading, not compiled, not run on hardware
//
#include <hip/hip_runtime.h>
#include <math.h>

#define N_NODES 30000
#define N_EDGES 300000
#define H 64
#define H3 192
#define NRBF 20
#define FIN 16
#define NVELS 5
#define CUTOFF 5.0f
#define EPSV 1e-8f

typedef __attribute__((ext_vector_type(16))) _Float16 v16h;
typedef __attribute__((ext_vector_type(8)))  _Float16 v8h;
typedef __attribute__((ext_vector_type(8)))  float    v8f;

__device__ __forceinline__ float silu_f(float t) { return t / (1.0f + expf(-t)); }
__device__ __forceinline__ float clip100(float t) { return fminf(fmaxf(t, -100.0f), 100.0f); }

// --- WMMA layouts per CDNA5 ISA 7.12.2 (wave32) ------------------------------
// A 16x32 f16: lane group g=lane/16, row m=lane%16; element e -> K:
//   e<8 : K = 8g+e (VGPR0..3) ; e>=8 : K = 16+8g+(e-8) (VGPR4..7)
__device__ __forceinline__ int a_kidx(int e, int g) {
  return (e < 8) ? (8 * g + e) : (16 + 8 * g + (e - 8));
}

__device__ __forceinline__ v16h load_a_f32(const float* __restrict__ src, int ld,
                                           int row0, int k0, int maxrow) {
  int lane = threadIdx.x & 31;
  int g = lane >> 4, m = lane & 15;
  int row = row0 + m; if (row > maxrow) row = maxrow;
  const float* p = src + (size_t)row * ld + k0;
  v16h a;
#pragma unroll
  for (int e = 0; e < 16; ++e) a[e] = (_Float16)p[a_kidx(e, g)];
  return a;
}

// Packed B tiles: tile = 32 lanes x 16 halves contiguous; lane loads 32B.
__device__ __forceinline__ v16h load_b_packed(const _Float16* __restrict__ Wp, int tile) {
  int lane = threadIdx.x & 31;
  return *(const v16h*)(Wp + ((size_t)tile * 32 + lane) * 16);
}

__device__ __forceinline__ v8f wmma32(v16h a, v16h b, v8f c) {
  return __builtin_amdgcn_wmma_f32_16x16x32_f16(false, a, false, b, (short)0, c,
                                                false, false);
}

// --- one-time weight prepack: f32 KxNc row-major -> f16 B-lane tiles ---------
// B 32x16 (KxN) lane layout: lane group g holds K = 8g+e (e<8), 16+8g+(e-8) (e>=8), col n = lane%16.
__global__ void pack_b_kernel(const float* __restrict__ W, int K, int Nc,
                              _Float16* __restrict__ dst) {
  int i = blockIdx.x * blockDim.x + threadIdx.x; // one per (tile,lane)
  int ntc = Nc >> 4;
  int total = (K >> 5) * ntc * 32;
  if (i >= total) return;
  int lane = i & 31;
  int tile = i >> 5;
  int kt = tile / ntc, ct = tile % ntc;
  int g = lane >> 4, n = lane & 15;
  const float* p = W + (size_t)(kt * 32 + 8 * g) * Nc + ct * 16 + n;
  _Float16* q = dst + (size_t)i * 16;
#pragma unroll
  for (int e = 0; e < 16; ++e) q[e] = (_Float16)p[(size_t)(e < 8 ? e : e + 8) * Nc];
}

// --- embed: s = feat@Wsemb + b ; v = vhist@Wvemb -----------------------------
__global__ void embed_kernel(const float* __restrict__ feat, const float* __restrict__ vhist,
                             const float* __restrict__ Ws, const float* __restrict__ bs,
                             const float* __restrict__ Wv,
                             float* __restrict__ s, float* __restrict__ v) {
  int i = blockIdx.x * blockDim.x + threadIdx.x;
  if (i >= N_NODES * H) return;
  int n = i >> 6, h = i & 63;
  float acc = bs[h];
#pragma unroll
  for (int k = 0; k < FIN; ++k) acc += feat[(size_t)n * FIN + k] * Ws[k * H + h];
  s[i] = acc;
#pragma unroll
  for (int d = 0; d < 3; ++d) {
    float a = 0.0f;
#pragma unroll
    for (int k = 0; k < NVELS; ++k)
      a += vhist[(size_t)n * 3 * NVELS + d * NVELS + k] * Wv[k * H + h];
    v[(size_t)n * H3 + d * H + h] = a;
  }
}

// --- interaction MLP: x = silu(s@W1+b1)@W2+b2 (WMMA, 4 waves/block) ----------
__global__ __launch_bounds__(128) void interact_kernel(
    const float* __restrict__ s, const _Float16* __restrict__ Wp1,
    const float* __restrict__ b1, const _Float16* __restrict__ Wp2,
    const float* __restrict__ b2, float* __restrict__ x) {
  __shared__ __align__(32) _Float16 hid[4][16 * H];
  int wave = threadIdx.x >> 5, lane = threadIdx.x & 31;
  int g = lane >> 4, n = lane & 15, m = lane & 15;
  int row0 = (blockIdx.x * 4 + wave) * 16;

  v8f acc[4] = {};
#pragma unroll
  for (int kt = 0; kt < 2; ++kt) {
    v16h a = load_a_f32(s, H, row0, kt * 32, N_NODES - 1);
#pragma unroll
    for (int ct = 0; ct < 4; ++ct)
      acc[ct] = wmma32(a, load_b_packed(Wp1, kt * 4 + ct), acc[ct]);
  }
#pragma unroll
  for (int ct = 0; ct < 4; ++ct)
#pragma unroll
    for (int r = 0; r < 8; ++r)
      hid[wave][(r + 8 * g) * H + ct * 16 + n] =
          (_Float16)silu_f(acc[ct][r] + b1[ct * 16 + n]);
  __syncthreads();

  v16h a0, a1;
  {
    const _Float16* hp = &hid[wave][m * H];
    v8h l0 = *(const v8h*)(hp + 8 * g);
    v8h h0 = *(const v8h*)(hp + 16 + 8 * g);
    v8h l1 = *(const v8h*)(hp + 32 + 8 * g);
    v8h h1 = *(const v8h*)(hp + 48 + 8 * g);
#pragma unroll
    for (int e = 0; e < 8; ++e) {
      a0[e] = l0[e]; a0[8 + e] = h0[e];
      a1[e] = l1[e]; a1[8 + e] = h1[e];
    }
  }
#pragma unroll
  for (int ct = 0; ct < 12; ++ct) {
    v8f c = {};
    c = wmma32(a0, load_b_packed(Wp2, ct), c);
    c = wmma32(a1, load_b_packed(Wp2, 12 + ct), c);
#pragma unroll
    for (int r = 0; r < 8; ++r) {
      int row = row0 + r + 8 * g;
      if (row < N_NODES) x[(size_t)row * H3 + ct * 16 + n] = c[r] + b2[ct * 16 + n];
    }
  }
}

// --- edge kernel: filter + message + scatter (one edge per wave32) -----------
__global__ void edge_kernel(const float* __restrict__ dist, const float* __restrict__ dir,
                            const int* __restrict__ snd, const int* __restrict__ rcv,
                            const float* __restrict__ Wf, const float* __restrict__ bf,
                            const float* __restrict__ x, const float* __restrict__ v,
                            float* __restrict__ dsa, float* __restrict__ dva) {
  int wave = threadIdx.x >> 5, lane = threadIdx.x & 31;
  int e = blockIdx.x * (blockDim.x >> 5) + wave;
  if (e >= N_EDGES) return;
  int si = snd[e], ri = rcv[e];
  // prefetch gathered rows (768B each) while we compute the RBF filter
  __builtin_prefetch(&x[(size_t)ri * H3 + lane * 6], 0, 0);
  __builtin_prefetch(&v[(size_t)ri * H3 + lane * 6], 0, 0);

  float dd = dist[e];
  float sg = CUTOFF / (NRBF - 1);
  float r = 0.0f;
  if (lane < NRBF) {
    float t = (dd - (float)lane * sg) / sg;
    r = expf(-0.5f * t * t);
  }
  float fc = (dd < CUTOFF) ? 0.5f * (cosf(3.14159265358979f * dd / CUTOFF) + 1.0f) : 0.0f;
  float d0 = dir[(size_t)e * 3 + 0], d1 = dir[(size_t)e * 3 + 1], d2 = dir[(size_t)e * 3 + 2];

  float wij[6];
#pragma unroll
  for (int j = 0; j < 6; ++j) wij[j] = bf[lane + 32 * j];
#pragma unroll
  for (int k = 0; k < NRBF; ++k) {
    float rk = __shfl(r, k, 32);
#pragma unroll
    for (int j = 0; j < 6; ++j) wij[j] += rk * Wf[k * H3 + lane + 32 * j];
  }
  float msg[6];
#pragma unroll
  for (int j = 0; j < 6; ++j)
    msg[j] = wij[j] * fc * x[(size_t)ri * H3 + lane + 32 * j];

  // cols 0..63 -> ds ; 64..127 -> dv1 ; 128..191 -> dv2
  atomicAdd(&dsa[(size_t)si * H + lane], msg[0]);
  atomicAdd(&dsa[(size_t)si * H + lane + 32], msg[1]);
  float dirv[3] = {d0, d1, d2};
#pragma unroll
  for (int d = 0; d < 3; ++d) {
    float vj0 = v[(size_t)ri * H3 + d * H + lane];
    float vj1 = v[(size_t)ri * H3 + d * H + lane + 32];
    atomicAdd(&dva[(size_t)si * H3 + d * H + lane], msg[2] * dirv[d] + msg[4] * vj0);
    atomicAdd(&dva[(size_t)si * H3 + d * H + lane + 32], msg[3] * dirv[d] + msg[5] * vj1);
  }
}

__global__ void apply1_kernel(float* __restrict__ s, float* __restrict__ v,
                              const float* __restrict__ dsa, const float* __restrict__ dva) {
  int i = blockIdx.x * blockDim.x + threadIdx.x;
  if (i >= N_NODES * H) return;
  int n = i >> 6, h = i & 63;
  s[i] += clip100(dsa[i]);
#pragma unroll
  for (int d = 0; d < 3; ++d) {
    size_t idx = (size_t)n * H3 + d * H + h;
    v[idx] += clip100(dva[idx]);
  }
}

// --- vw = v(3N x 64) @ Wvm(64 x 128) (WMMA) ----------------------------------
__global__ __launch_bounds__(128) void vw_kernel(const float* __restrict__ v,
                                                 const _Float16* __restrict__ Wp,
                                                 float* __restrict__ vw) {
  int wave = threadIdx.x >> 5, lane = threadIdx.x & 31;
  int g = lane >> 4, n = lane & 15;
  int row0 = (blockIdx.x * 4 + wave) * 16;
  const int M = 3 * N_NODES;
  v16h a0 = load_a_f32(v, H, row0, 0, M - 1);
  v16h a1 = load_a_f32(v, H, row0, 32, M - 1);
#pragma unroll
  for (int ct = 0; ct < 8; ++ct) {
    v8f c = {};
    c = wmma32(a0, load_b_packed(Wp, ct), c);
    c = wmma32(a1, load_b_packed(Wp, 8 + ct), c);
#pragma unroll
    for (int r = 0; r < 8; ++r) {
      int row = row0 + r + 8 * g;
      if (row < M) vw[(size_t)row * 128 + ct * 16 + n] = c[r];
    }
  }
}

__global__ void normdot_kernel(const float* __restrict__ vw, float* __restrict__ vnorm,
                               float* __restrict__ vdot) {
  int i = blockIdx.x * blockDim.x + threadIdx.x;
  if (i >= N_NODES * H) return;
  int n = i >> 6, h = i & 63;
  float nn = EPSV, dt = 0.0f;
#pragma unroll
  for (int d = 0; d < 3; ++d) {
    float vl = vw[(size_t)(3 * n + d) * 128 + h];
    float vr = vw[(size_t)(3 * n + d) * 128 + 64 + h];
    nn += vr * vr;
    dt += vr * vl;
  }
  vnorm[i] = sqrtf(nn);
  vdot[i] = dt;
}

// --- mix = silu([s|vnorm]@Wm1+b)@Wm2+b  (WMMA, K=128 composite A) ------------
__global__ __launch_bounds__(128) void mix_kernel(
    const float* __restrict__ s, const float* __restrict__ vnorm,
    const _Float16* __restrict__ Wp1, const float* __restrict__ b1,
    const _Float16* __restrict__ Wp2, const float* __restrict__ b2,
    float* __restrict__ mix) {
  __shared__ __align__(32) _Float16 hid[4][16 * H];
  int wave = threadIdx.x >> 5, lane = threadIdx.x & 31;
  int g = lane >> 4, n = lane & 15, m = lane & 15;
  int row0 = (blockIdx.x * 4 + wave) * 16;
  int row = row0 + m; if (row > N_NODES - 1) row = N_NODES - 1;

  v8f acc[4] = {};
#pragma unroll
  for (int kt = 0; kt < 4; ++kt) {
    v16h a;
#pragma unroll
    for (int e = 0; e < 16; ++e) {
      int k = kt * 32 + a_kidx(e, g);
      a[e] = (_Float16)((k < H) ? s[(size_t)row * H + k] : vnorm[(size_t)row * H + (k - H)]);
    }
#pragma unroll
    for (int ct = 0; ct < 4; ++ct)
      acc[ct] = wmma32(a, load_b_packed(Wp1, kt * 4 + ct), acc[ct]);
  }
#pragma unroll
  for (int ct = 0; ct < 4; ++ct)
#pragma unroll
    for (int r = 0; r < 8; ++r)
      hid[wave][(r + 8 * g) * H + ct * 16 + n] =
          (_Float16)silu_f(acc[ct][r] + b1[ct * 16 + n]);
  __syncthreads();

  v16h a0, a1;
  {
    const _Float16* hp = &hid[wave][m * H];
    v8h l0 = *(const v8h*)(hp + 8 * g);
    v8h h0 = *(const v8h*)(hp + 16 + 8 * g);
    v8h l1 = *(const v8h*)(hp + 32 + 8 * g);
    v8h h1 = *(const v8h*)(hp + 48 + 8 * g);
#pragma unroll
    for (int e = 0; e < 8; ++e) {
      a0[e] = l0[e]; a0[8 + e] = h0[e];
      a1[e] = l1[e]; a1[8 + e] = h1[e];
    }
  }
#pragma unroll
  for (int ct = 0; ct < 12; ++ct) {
    v8f c = {};
    c = wmma32(a0, load_b_packed(Wp2, ct), c);
    c = wmma32(a1, load_b_packed(Wp2, 12 + ct), c);
#pragma unroll
    for (int r = 0; r < 8; ++r) {
      int rw = row0 + r + 8 * g;
      if (rw < N_NODES) mix[(size_t)rw * H3 + ct * 16 + n] = c[r] + b2[ct * 16 + n];
    }
  }
}

__global__ void apply2_kernel(float* __restrict__ s, float* __restrict__ v,
                              const float* __restrict__ mix, const float* __restrict__ vw,
                              const float* __restrict__ vdot) {
  int i = blockIdx.x * blockDim.x + threadIdx.x;
  if (i >= N_NODES * H) return;
  int n = i >> 6, h = i & 63;
  float ds = mix[(size_t)n * H3 + h];
  float dvv = mix[(size_t)n * H3 + H + h];
  float dsv = mix[(size_t)n * H3 + 2 * H + h];
  s[i] += clip100(ds + dsv * vdot[i]);
#pragma unroll
  for (int d = 0; d < 3; ++d)
    v[(size_t)n * H3 + d * H + h] += clip100(vw[(size_t)(3 * n + d) * 128 + h] * dvv);
}

// --- gated equivariant block 0 (H=64 -> 32) ----------------------------------
__global__ void gb1_vw_kernel(const float* __restrict__ v, const float* __restrict__ W,
                              float* __restrict__ vw1) {
  int i = blockIdx.x * blockDim.x + threadIdx.x;
  if (i >= N_NODES * H3) return;
  int n = i / H3, c = i % H3, d = c >> 6, j = c & 63;
  float a = 0.0f;
#pragma unroll
  for (int k = 0; k < H; ++k) a += v[(size_t)n * H3 + d * H + k] * W[k * H + j];
  vw1[i] = a;
}

__global__ void gb1_norm_kernel(const float* __restrict__ vw1, float* __restrict__ nrm) {
  int i = blockIdx.x * blockDim.x + threadIdx.x;
  if (i >= N_NODES * 32) return;
  int n = i >> 5, j = i & 31;
  float nn = EPSV;
#pragma unroll
  for (int d = 0; d < 3; ++d) {
    float t = vw1[(size_t)n * H3 + d * H + 32 + j];
    nn += t * t;
  }
  nrm[i] = sqrtf(nn);
}

__global__ void gb1_h_kernel(const float* __restrict__ s, const float* __restrict__ nrm,
                             const float* __restrict__ Wg1, const float* __restrict__ bg1,
                             float* __restrict__ hid) {
  int i = blockIdx.x * blockDim.x + threadIdx.x;
  if (i >= N_NODES * H) return;
  int n = i >> 6, o = i & 63;
  float t = bg1[o];
#pragma unroll
  for (int k = 0; k < H; ++k) t += s[(size_t)n * H + k] * Wg1[k * H + o];
#pragma unroll
  for (int k = 0; k < 32; ++k) t += nrm[(size_t)n * 32 + k] * Wg1[(H + k) * H + o];
  hid[i] = silu_f(t);
}

__global__ void gb1_g_kernel(const float* __restrict__ hid, const float* __restrict__ Wg2,
                             const float* __restrict__ bg2, float* __restrict__ g) {
  int i = blockIdx.x * blockDim.x + threadIdx.x;
  if (i >= N_NODES * H) return;
  int n = i >> 6, o = i & 63;
  float t = bg2[o];
#pragma unroll
  for (int k = 0; k < H; ++k) t += hid[(size_t)n * H + k] * Wg2[k * H + o];
  g[i] = t;
}

__global__ void gb1_out_kernel(const float* __restrict__ g, const float* __restrict__ vw1,
                               float* __restrict__ s2, float* __restrict__ v2) {
  int i = blockIdx.x * blockDim.x + threadIdx.x;
  if (i >= N_NODES * 32) return;
  int n = i >> 5, j = i & 31;
  s2[i] = g[(size_t)n * H + j];
  float gate = g[(size_t)n * H + 32 + j];
#pragma unroll
  for (int d = 0; d < 3; ++d)
    v2[(size_t)n * 96 + d * 32 + j] = vw1[(size_t)n * H3 + d * H + j] * gate;
}

// --- gated equivariant block 1 (32 -> 1), writes final outputs ---------------
__global__ void gb2_kernel(const float* __restrict__ s2, const float* __restrict__ v2,
                           const float* __restrict__ Wvm, const float* __restrict__ Wg1,
                           const float* __restrict__ bg1, const float* __restrict__ Wg2,
                           const float* __restrict__ bg2, float* __restrict__ out) {
  int n = blockIdx.x * blockDim.x + threadIdx.x;
  if (n >= N_NODES) return;
  float vl[3], vr[3];
#pragma unroll
  for (int d = 0; d < 3; ++d) {
    float a = 0.0f, b = 0.0f;
#pragma unroll
    for (int j = 0; j < 32; ++j) {
      float t = v2[(size_t)n * 96 + d * 32 + j];
      a += t * Wvm[j * 2 + 0];
      b += t * Wvm[j * 2 + 1];
    }
    vl[d] = a; vr[d] = b;
  }
  float nrm = sqrtf(vr[0] * vr[0] + vr[1] * vr[1] + vr[2] * vr[2] + EPSV);
  float hid[32];
#pragma unroll
  for (int o = 0; o < 32; ++o) {
    float t = bg1[o];
#pragma unroll
    for (int k = 0; k < 32; ++k) t += s2[(size_t)n * 32 + k] * Wg1[k * 32 + o];
    t += nrm * Wg1[32 * 32 + o];
    hid[o] = silu_f(t);
  }
  float g0 = bg2[0], g1 = bg2[1];
#pragma unroll
  for (int k = 0; k < 32; ++k) {
    g0 += hid[k] * Wg2[k * 2 + 0];
    g1 += hid[k] * Wg2[k * 2 + 1];
  }
  out[n] = g0;
#pragma unroll
  for (int d = 0; d < 3; ++d) out[N_NODES + (size_t)n * 3 + d] = vl[d] * g1;
}

extern "C" void kernel_launch(void* const* d_in, const int* in_sizes, int n_in,
                              void* d_out, int out_size, void* d_ws, size_t ws_size,
                              hipStream_t stream) {
  const float* feat    = (const float*)d_in[0];
  const float* vhist   = (const float*)d_in[1];
  const float* dist    = (const float*)d_in[2];
  const float* dir     = (const float*)d_in[3];
  const float* Wsemb   = (const float*)d_in[4];
  const float* bsemb   = (const float*)d_in[5];
  const float* Wvemb   = (const float*)d_in[6];
  const float* Wfilt   = (const float*)d_in[7];
  const float* bfilt   = (const float*)d_in[8];
  const float* Wi1     = (const float*)d_in[9];
  const float* bi1     = (const float*)d_in[10];
  const float* Wi2     = (const float*)d_in[11];
  const float* bi2     = (const float*)d_in[12];
  const float* Wm1     = (const float*)d_in[13];
  const float* bm1     = (const float*)d_in[14];
  const float* Wm2     = (const float*)d_in[15];
  const float* bm2     = (const float*)d_in[16];
  const float* Wvm     = (const float*)d_in[17];
  const float* r0Wvm   = (const float*)d_in[18];
  const float* r0Wg1   = (const float*)d_in[19];
  const float* r0bg1   = (const float*)d_in[20];
  const float* r0Wg2   = (const float*)d_in[21];
  const float* r0bg2   = (const float*)d_in[22];
  const float* roWvm   = (const float*)d_in[23];
  const float* roWg1   = (const float*)d_in[24];
  const float* robg1   = (const float*)d_in[25];
  const float* roWg2   = (const float*)d_in[26];
  const float* robg2   = (const float*)d_in[27];
  const int* senders   = (const int*)d_in[28];
  const int* receivers = (const int*)d_in[29];

  float* ws = (float*)d_ws;
  const size_t NN = N_NODES;
  float* s     = ws;                 // N*64
  float* v     = ws + NN * 64;       // N*192
  float* x     = ws + NN * 256;      // N*192 (reused as mix)
  float* dsa   = ws + NN * 448;      // N*64  (reused as gb1 hid)
  float* dva   = ws + NN * 512;      // N*192 (reused as gb1 g [N*64] + v2 [N*96])
  float* vw    = ws + NN * 704;      // 3N*128 (reused as gb1 vw1)
  float* vnorm = ws + NN * 1088;     // N*64  (reused as gb1 nrm)
  float* vdot  = ws + NN * 1152;     // N*64  (reused as gb1 s2)
  float* hid1  = dsa;
  float* g1    = dva;
  float* v2    = dva + NN * 64;
  float* s2    = vdot;

  // f16 packed-weight region (per layer: Wi1 4096, Wi2 12288, Wm1 8192,
  // Wm2 12288, Wvm 8192 halves = 45056)
  _Float16* wp = (_Float16*)(ws + NN * 1216);
  const size_t WPL = 45056;
  _Float16* wpWi1[2] = { wp,               wp + WPL };
  _Float16* wpWi2[2] = { wp + 4096,        wp + WPL + 4096 };
  _Float16* wpWm1[2] = { wp + 16384,       wp + WPL + 16384 };
  _Float16* wpWm2[2] = { wp + 24576,       wp + WPL + 24576 };
  _Float16* wpWvm[2] = { wp + 36864,       wp + WPL + 36864 };

  const int TPB = 256;
  const int gNH    = (N_NODES * H + TPB - 1) / TPB;
  const int gN32   = (N_NODES * 32 + TPB - 1) / TPB;
  const int gN192  = (N_NODES * H3 + TPB - 1) / TPB;
  const int gWM_N  = (N_NODES + 63) / 64;              // 4 waves x 16 rows
  const int gWM_3N = (3 * N_NODES + 63) / 64;
  const int gEdge  = (N_EDGES + 7) / 8;                // 8 waves/block
  const int gNode  = (N_NODES + TPB - 1) / TPB;

  // prepack all WMMA weights to f16 B-lane tiles (tiny, runs every call)
  for (int l = 0; l < 2; ++l) {
    pack_b_kernel<<<1, 256, 0, stream>>>(Wi1 + l * H * H, H, H, wpWi1[l]);
    pack_b_kernel<<<3, 256, 0, stream>>>(Wi2 + l * H * H3, H, H3, wpWi2[l]);
    pack_b_kernel<<<2, 256, 0, stream>>>(Wm1 + l * 2 * H * H, 2 * H, H, wpWm1[l]);
    pack_b_kernel<<<3, 256, 0, stream>>>(Wm2 + l * H * H3, H, H3, wpWm2[l]);
    pack_b_kernel<<<2, 256, 0, stream>>>(Wvm + l * H * 2 * H, H, 2 * H, wpWvm[l]);
  }

  embed_kernel<<<gNH, TPB, 0, stream>>>(feat, vhist, Wsemb, bsemb, Wvemb, s, v);

  for (int l = 0; l < 2; ++l) {
    interact_kernel<<<gWM_N, 128, 0, stream>>>(s, wpWi1[l], bi1 + l * H,
                                               wpWi2[l], bi2 + l * H3, x);
    hipMemsetAsync(dsa, 0, NN * 256 * sizeof(float), stream);
    edge_kernel<<<gEdge, TPB, 0, stream>>>(dist, dir, senders, receivers, Wfilt, bfilt,
                                           x, v, dsa, dva);
    apply1_kernel<<<gNH, TPB, 0, stream>>>(s, v, dsa, dva);
    vw_kernel<<<gWM_3N, 128, 0, stream>>>(v, wpWvm[l], vw);
    normdot_kernel<<<gNH, TPB, 0, stream>>>(vw, vnorm, vdot);
    mix_kernel<<<gWM_N, 128, 0, stream>>>(s, vnorm, wpWm1[l], bm1 + l * H,
                                          wpWm2[l], bm2 + l * H3, x);
    apply2_kernel<<<gNH, TPB, 0, stream>>>(s, v, x, vw, vdot);
  }

  gb1_vw_kernel<<<gN192, TPB, 0, stream>>>(v, r0Wvm, vw);
  gb1_norm_kernel<<<gN32, TPB, 0, stream>>>(vw, vnorm);
  gb1_h_kernel<<<gNH, TPB, 0, stream>>>(s, vnorm, r0Wg1, r0bg1, hid1);
  gb1_g_kernel<<<gNH, TPB, 0, stream>>>(hid1, r0Wg2, r0bg2, g1);
  gb1_out_kernel<<<gN32, TPB, 0, stream>>>(g1, vw, s2, v2);
  gb2_kernel<<<gNode, TPB, 0, stream>>>(s2, v2, roWvm, roWg1, robg1, roWg2, robg2,
                                        (float*)d_out);
}
